// deconv2d_34918084116993
// MI455X (gfx1250) — matrix operand order repaired and checked by reading, not compiled, not run on hardware
//
#include <hip/hip_runtime.h>
#include <hip/hip_bf16.h>

typedef __bf16 bf16;
typedef __attribute__((ext_vector_type(16))) __bf16 v16bf;
typedef __attribute__((ext_vector_type(8)))  float  v8f;

#define BB    2
#define HIN   32
#define WIN   128
#define CIN   128
#define HO    64
#define WO    256
#define HP    66
#define WP    258
#define KK    9
#define FO    256
#define KTOT  1152          // 9 * 128
#define BM    128           // rows per block
#define MBLK  256           // 32768 / 128

// ---------------------------------------------------------------------------
// Kernel 1: bilinear resize (half-pixel centers, scale 2) + SAME pad -> ws
// ---------------------------------------------------------------------------
__global__ __launch_bounds__(128) void prep_image(const float* __restrict__ x,
                                                  float* __restrict__ imP) {
  const int c   = threadIdx.x;          // channel
  int idx = blockIdx.x;                 // b * HP * WP
  const int wp = idx % WP; idx /= WP;
  const int hp = idx % HP;
  const int b  = idx / HP;
  float v = 0.0f;
  if (hp >= 1 && hp <= HO && wp >= 1 && wp <= WO) {
    const int h = hp - 1, w = wp - 1;
    const float sy = h * 0.5f - 0.25f;
    const float sx = w * 0.5f - 0.25f;
    const float y0f = floorf(sy), x0f = floorf(sx);
    const float ty = sy - y0f, tx = sx - x0f;
    const int iy0 = min(max((int)y0f,     0), HIN - 1);
    const int iy1 = min(max((int)y0f + 1, 0), HIN - 1);
    const int ix0 = min(max((int)x0f,     0), WIN - 1);
    const int ix1 = min(max((int)x0f + 1, 0), WIN - 1);
    const float* xb = x + (size_t)b * HIN * WIN * CIN;
    const float p00 = xb[(iy0 * WIN + ix0) * CIN + c];
    const float p01 = xb[(iy0 * WIN + ix1) * CIN + c];
    const float p10 = xb[(iy1 * WIN + ix0) * CIN + c];
    const float p11 = xb[(iy1 * WIN + ix1) * CIN + c];
    v = (1.f - ty) * ((1.f - tx) * p00 + tx * p01)
      +        ty  * ((1.f - tx) * p10 + tx * p11);
  }
  imP[((size_t)(b * HP + hp) * WP + wp) * CIN + c] = v;
}

// ---------------------------------------------------------------------------
// Kernel 2: pack f32 weights -> bf16 in WMMA B-fragment order.
// B 32x16 bf16 order used here (contiguous k): lanes 0-15 K=0..15, lanes 16-31 K=16..31,
// 16 bf16 per lane.  Linearized: ((kblock*16 + ntile)*512) + lane*16 + e.
// ---------------------------------------------------------------------------
__global__ __launch_bounds__(256) void pack_w(const float* __restrict__ kw,
                                              bf16* __restrict__ wpk) {
  const int idx = blockIdx.x * 256 + threadIdx.x;   // < 1152*256
  const int k = idx >> 8;                           // K row 0..1151
  const int n = idx & 255;                          // col 0..255
  const int kbg = k >> 5;                           // K-block of 32
  const int K   = k & 31;
  const int nt  = n >> 4;                           // N tile
  const int col = n & 15;
  const int laneHi = (K >> 4) & 1;
  const int e      = K & 15;
  const int lane   = laneHi * 16 + col;
  wpk[((kbg * 16 + nt) * 512) + lane * 16 + e] = (bf16)kw[idx];
}

// ---------------------------------------------------------------------------
// Kernel 3: fused deformable gather + implicit GEMM via v_wmma_f32_16x16x32_bf16
// ---------------------------------------------------------------------------
struct RowCoord { unsigned o[4]; float w[4]; };

__device__ inline float4 fma4(float w, float4 v, float4 a) {
  a.x += w * v.x; a.y += w * v.y; a.z += w * v.z; a.w += w * v.w; return a;
}

__global__ __launch_bounds__(256) void fused_gemm(const float* __restrict__ imP,
                                                  const bf16*  __restrict__ wpk,
                                                  const float* __restrict__ bias,
                                                  const float* __restrict__ off,
                                                  float* __restrict__ out) {
  __shared__ __align__(32) bf16 Asm[BM * CIN];   // 32 KB, A fragments for one tap
  __shared__ RowCoord rc[BM];                    // 4 KB
  const int tid  = threadIdx.x;
  const int lane = tid & 31;
  const int mt   = tid >> 5;                     // wave id 0..7 -> M sub-tile
  const int m0   = blockIdx.x * BM;

  v8f acc[16];
  #pragma unroll
  for (int nt = 0; nt < 16; ++nt) acc[nt] = (v8f){0,0,0,0,0,0,0,0};

  for (int j = 0; j < KK; ++j) {
    const int iyj = j / 3, ixj = j % 3;          // extract_patches row-major
    // ---- per-row sampling coords (offsets depend only on (h, tap)) ----
    if (tid < BM) {
      const int m  = m0 + tid;
      const int b  = m >> 14;                    // / (64*256)
      const int hw = m & 16383;
      const int h  = hw >> 8;
      const int w  = hw & 255;
      const float offy = off[((h * WO) * KK + j) * 2 + 0];
      const float offx = off[((h * WO) * KK + j) * 2 + 1];
      float y = (float)(h + iyj) + offy;
      y = fminf(fmaxf(y, 0.f), (float)(HP - 1));
      int y0 = min(max((int)floorf(y), 0), HP - 1);
      int y1 = min(y0 + 1, HP - 1);
      const float wy1 = y - (float)y0;
      const float wy0 = (float)y1 - y;
      float xc = (float)(w + ixj) + offx;
      if (xc < 0.f)              xc += (float)WP;   // wrap once, like reference
      if (xc > (float)(WP - 1))  xc -= (float)WP;
      const int x0 = (int)floorf(xc);
      const int x1 = x0 + 1;
      const float wx1 = xc - (float)x0;             // weights use unwrapped coords
      const float wx0 = (float)x1 - xc;
      const int x0i = x0 < 0 ? x0 + WP : (x0 > WP - 1 ? x0 - WP : x0);
      const int x1i = x1 < 0 ? x1 + WP : (x1 > WP - 1 ? x1 - WP : x1);
      const unsigned b0 = (unsigned)((b * HP + y0) * WP);
      const unsigned b1 = (unsigned)((b * HP + y1) * WP);
      rc[tid].o[0] = (b0 + x0i) * CIN;
      rc[tid].o[1] = (b0 + x1i) * CIN;
      rc[tid].o[2] = (b1 + x0i) * CIN;
      rc[tid].o[3] = (b1 + x1i) * CIN;
      rc[tid].w[0] = wy0 * wx0;
      rc[tid].w[1] = wy0 * wx1;
      rc[tid].w[2] = wy1 * wx0;
      rc[tid].w[3] = wy1 * wx1;
    }
    __syncthreads();

    // ---- stage 128 rows x 128 ch of bilinear-gathered bf16, fragment order ----
    // 8 consecutive channels (c%8==0) -> same fragment lane, 8 consecutive e:
    //   K=c&31 in {0,8,16,24} -> laneHi=(K>>4)&1, e0 in {0,8,0,8}; runs of 8.
    // So each thread: 8x global_load_b128 + one 16B ds_store_b128.
    #pragma unroll 2
    for (int i = 0; i < (BM * CIN / 8) / 256; ++i) {   // 8 iters
      const int el = i * 256 + tid;     // 8-channel group index
      const int r  = el >> 4;           // 16 groups per row
      const int g  = el & 15;
      const int c  = g * 8;
      const RowCoord& q = rc[r];
      float4 s0 = {0.f, 0.f, 0.f, 0.f};
      float4 s1 = {0.f, 0.f, 0.f, 0.f};
      #pragma unroll
      for (int n = 0; n < 4; ++n) {
        const float4* pn = reinterpret_cast<const float4*>(imP + q.o[n] + c);
        s0 = fma4(q.w[n], pn[0], s0);
        s1 = fma4(q.w[n], pn[1], s1);
      }
      const int kb     = c >> 5;
      const int K      = c & 31;
      const int laneHi = (K >> 4) & 1;
      const int Klow   = K - laneHi * 16;
      const int e0     = Klow & 15;
      const int mtt    = r >> 4;
      const int idx    = ((mtt * 4 + kb) * 512) + (laneHi * 16 + (r & 15)) * 16 + e0;
      union { bf16 h[8]; uint4 u; } pk;
      pk.h[0] = (bf16)s0.x; pk.h[1] = (bf16)s0.y;
      pk.h[2] = (bf16)s0.z; pk.h[3] = (bf16)s0.w;
      pk.h[4] = (bf16)s1.x; pk.h[5] = (bf16)s1.y;
      pk.h[6] = (bf16)s1.z; pk.h[7] = (bf16)s1.w;
      *reinterpret_cast<uint4*>(&Asm[idx]) = pk.u;   // 16B aligned: idx%8==0
    }
    __syncthreads();

    // ---- 4 K-steps x 16 N-tiles of WMMA; A from LDS, B from L2-resident pack ----
    #pragma unroll
    for (int kb = 0; kb < 4; ++kb) {
      const v16bf a = *reinterpret_cast<const v16bf*>(&Asm[((mt * 4 + kb) * 512) + lane * 16]);
      const bf16* bp = wpk + ((size_t)(j * 4 + kb) * 16) * 512 + lane * 16;
      #pragma unroll
      for (int nt = 0; nt < 16; ++nt) {
        const v16bf bfr = *reinterpret_cast<const v16bf*>(bp + nt * 512);
        acc[nt] = __builtin_amdgcn_wmma_f32_16x16x32_bf16(
            false, a, false, bfr, (short)0, acc[nt], false, false);
      }
    }
    __syncthreads();
  }

  // ---- epilogue: C/D layout VGPR r -> M = r + 8*(lane>=16), N = lane&15 ----
  const int laneHi = lane >> 4;
  const int l15    = lane & 15;
  #pragma unroll
  for (int nt = 0; nt < 16; ++nt) {
    const int coln = nt * 16 + l15;
    const float bz = bias[coln];
    #pragma unroll
    for (int r = 0; r < 8; ++r) {
      const int row = m0 + mt * 16 + r + laneHi * 8;
      __builtin_nontemporal_store(acc[nt][r] + bz, &out[(size_t)row * FO + coln]);
    }
  }
}

// ---------------------------------------------------------------------------
extern "C" void kernel_launch(void* const* d_in, const int* in_sizes, int n_in,
                              void* d_out, int out_size, void* d_ws, size_t ws_size,
                              hipStream_t stream) {
  (void)in_sizes; (void)n_in; (void)out_size; (void)ws_size;
  const float* x    = (const float*)d_in[0];
  const float* kern = (const float*)d_in[1];
  const float* bias = (const float*)d_in[2];
  const float* off  = (const float*)d_in[3];
  float* out = (float*)d_out;

  float* imP = (float*)d_ws;                                   // 17.4 MB
  bf16*  wpk = (bf16*)((char*)d_ws +
                       (size_t)BB * HP * WP * CIN * sizeof(float)); // +0.59 MB

  prep_image<<<dim3(BB * HP * WP), 128, 0, stream>>>(x, imP);
  pack_w<<<dim3((KTOT * FO) / 256), 256, 0, stream>>>(kern, wpk);
  fused_gemm<<<dim3(MBLK), 256, 0, stream>>>(imP, wpk, bias, off, out);
}
